// MultiTokenPrediction_14594298871876
// MI455X (gfx1250) — matrix operand order repaired and checked by reading, not compiled, not run on hardware
//
#include <hip/hip_runtime.h>
#include <hip/hip_bf16.h>
#include <math.h>

// ---------------------------------------------------------------------------
// MultiTokenPrediction for MI455X (gfx1250, wave32, WMMA)
//   hidden (1,2048,1024) f32 ; targets (2048) ; emb_w (V,1024)
//   aux_proj_w (2,1024,1024) ; aux_norm_w (2,1024) ; aux_out_w (2,V,1024)
//   out: main_logits (2048*V) f32  ++  loss (1) f32
//
// Roofline: 3 vocab GEMMs (2048x1024x50257) ~ 632 GFLOP, weights 617 MB.
// CDNA5 lever: 320 KB LDS/WGP -> the whole 128x1024 weight slab (bf16,
// padded row stride for conflict-free b128 fragment reads) stays resident in
// LDS while the workgroup loops M-blocks over it => weights stream from HBM
// ~2x (~53 us) instead of 16x. K-loop is barrier-free. An empty asm memory
// clobber at the top of each M-block iteration blocks LICM from hoisting the
// (mb-invariant) LDS fragment loads out of the loop (round-2 failure:
// 8 KB/lane hoisted + spilled to scratch) while keeping them ordinary AS(3)
// loads -> ds_load_b128 with backend-managed s_wait_dscnt (round-3 failure:
// volatile generic pointers became system-scope flat_loads).
// Aux logits (2 x 412 MB) are never materialized: fused streaming
// cross-entropy (half-wave shuffles + global_atomic_add_f32).
// ---------------------------------------------------------------------------

typedef __attribute__((ext_vector_type(16))) __bf16      v16bf;
typedef __attribute__((ext_vector_type(8)))  float       v8f;
typedef __attribute__((ext_vector_type(4)))  unsigned    uint4v;

#define S_LEN   2048
#define D_DIM   1024
#define V_DIM   50257
#define EPSV    1e-5f
#define LOSS_W  0.3f
#define IGNORE_IDX (-100)

// padded LDS row stride: 516 dwords, 516 % 64 == 4 -> a 16-lane b128
// fragment read covers all 64 LDS banks exactly once (conflict-free)
#define LDW     (D_DIM + 8)
#define LDS_BYTES ((size_t)128 * LDW * sizeof(__bf16))   // 264192 B < 320 KB

// ---------------------------------------------------------------------------
__global__ void init_stats_kernel(float* __restrict__ sumexp,
                                  float* __restrict__ tgtl, int n) {
  int i = blockIdx.x * blockDim.x + threadIdx.x;
  if (i < n) { sumexp[i] = 0.0f; tgtl[i] = 0.0f; }
}

// ---------------------------------------------------------------------------
// Row-wise RMSNorm (f32 in) -> bf16 normalized out; optional raw bf16 copy.
// ---------------------------------------------------------------------------
__global__ __launch_bounds__(256)
void rmsnorm_bf16_kernel(const float* __restrict__ X,
                         const float* __restrict__ w,
                         __bf16* __restrict__ Aout,
                         __bf16* __restrict__ rawOut) {
  const int s = blockIdx.x;
  const float* x = X + (size_t)s * D_DIM;

  float ss = 0.0f;
  for (int d = threadIdx.x; d < D_DIM; d += blockDim.x) {
    float v = x[d];
    ss += v * v;
  }
  #pragma unroll
  for (int m = 16; m >= 1; m >>= 1) ss += __shfl_xor(ss, m, 32);
  __shared__ float red[8];
  const int wv = threadIdx.x >> 5, ln = threadIdx.x & 31;
  if (ln == 0) red[wv] = ss;
  __syncthreads();
  if (threadIdx.x == 0) {
    float a = 0.0f;
    #pragma unroll
    for (int i = 0; i < 8; ++i) a += red[i];
    red[0] = a;
  }
  __syncthreads();
  const float rs = rsqrtf(red[0] / (float)D_DIM + EPSV);

  for (int d = threadIdx.x; d < D_DIM; d += blockDim.x) {
    float v = x[d];
    Aout[(size_t)s * D_DIM + d] = (__bf16)(v * rs * w[d]);
    if (rawOut) rawOut[(size_t)s * D_DIM + d] = (__bf16)v;
  }
}

// ---------------------------------------------------------------------------
// LDS-resident-weight WMMA GEMM:  C[m,n] = sum_k A[m,k] * W[n,k]
//   Phase 1: stage 128 N-rows x 1024 K of W (fp32->bf16) into dynamic LDS.
//   Phase 2: loop M-blocks (stride gridDim.y); each of 8 waves owns a
//            16x128 strip = 8 accumulators; K-loop: prefetch next A frag
//            (global, L2-hot), load 8 B frags (ds_load_b128), 8 WMMAs.
//            No barriers after the single __syncthreads.
//   Epilogue: optional f32 store; optional fused CE stats.
// ---------------------------------------------------------------------------
template <bool STORE_F32, bool STATS>
__global__ __launch_bounds__(256)
void gemm_wslab_wmma_kernel(const __bf16* __restrict__ A,   // S x 1024 bf16
                            const float*  __restrict__ W,   // Nrows x 1024 f32
                            int Nrows,
                            float* __restrict__ Out, long long ldOut,
                            float* __restrict__ sumexp,
                            float* __restrict__ tgtLogit,
                            const int* __restrict__ targets,
                            int shiftK) {
  constexpr int K = D_DIM;
  extern __shared__ __bf16 lW[];                 // [128][LDW]

  const int n0   = blockIdx.x * 128;
  const int tid  = threadIdx.x;
  const int wave = tid >> 5;
  const int lane = tid & 31;
  const int lhalf = lane >> 4;     // K-half selector per WMMA operand layout
  const int l16   = lane & 15;

  // ---- Phase 1: stage + convert the whole 128 x 1024 weight slab ----
  {
    const int row  = tid >> 1;             // 0..127
    const int cb   = (tid & 1) * (K / 2);  // 0 or 512
    const int wrow = n0 + row;
    for (int c = 0; c < K / 2; c += 16) {
      float4 f0 = {0,0,0,0}, f1 = {0,0,0,0}, f2 = {0,0,0,0}, f3 = {0,0,0,0};
      if (wrow < Nrows) {
        const float* src = W + (size_t)wrow * K + cb + c;
        f0 = ((const float4*)src)[0];
        f1 = ((const float4*)src)[1];
        f2 = ((const float4*)src)[2];
        f3 = ((const float4*)src)[3];
      }
      union { __bf16 h[16]; uint4v u[2]; } pk;
      pk.h[0]=(__bf16)f0.x; pk.h[1]=(__bf16)f0.y; pk.h[2]=(__bf16)f0.z; pk.h[3]=(__bf16)f0.w;
      pk.h[4]=(__bf16)f1.x; pk.h[5]=(__bf16)f1.y; pk.h[6]=(__bf16)f1.z; pk.h[7]=(__bf16)f1.w;
      pk.h[8]=(__bf16)f2.x; pk.h[9]=(__bf16)f2.y; pk.h[10]=(__bf16)f2.z; pk.h[11]=(__bf16)f2.w;
      pk.h[12]=(__bf16)f3.x; pk.h[13]=(__bf16)f3.y; pk.h[14]=(__bf16)f3.z; pk.h[15]=(__bf16)f3.w;
      uint4v* dst = (uint4v*)&lW[(size_t)row * LDW + cb + c];
      dst[0] = pk.u[0];
      dst[1] = pk.u[1];
    }
  }
  __syncthreads();   // the ONLY barrier: W slab is read-only afterwards

  // ---- Phase 2: M-blocks over the resident weight slab ----
  for (int mb = blockIdx.y; mb < S_LEN / 128; mb += gridDim.y) {
    // May-store to any location: no load (in particular the mb-invariant
    // LDS fragment loads below) can be hoisted across this point, so the
    // ds_load_b128s stay inside the loop without becoming volatile/flat.
    asm volatile("" ::: "memory");

    const int mwbase = mb * 128 + wave * 16;
    const __bf16* arow = A + (size_t)(mwbase + l16) * K;

    v8f acc[8];
    #pragma unroll
    for (int i = 0; i < 8; ++i)
      #pragma unroll
      for (int j = 0; j < 8; ++j) acc[i][j] = 0.0f;

    // A fragment: 16-bit A 16x32 layout; lane half picks K {0..7,16..23}
    // (half 0) or {8..15,24..31} (half 1)
    union { uint4v u[2]; v16bf v; } afc, afn;
    afc.u[0] = *(const uint4v*)(arow + lhalf * 8);
    afc.u[1] = *(const uint4v*)(arow + 16 + lhalf * 8);

    for (int kt = 0; kt < K; kt += 32) {
      // prefetch next K-step's A fragment (overlaps WMMA stream)
      if (kt + 32 < K) {
        afn.u[0] = *(const uint4v*)(arow + (kt + 32) + lhalf * 8);
        afn.u[1] = *(const uint4v*)(arow + (kt + 48) + lhalf * 8);
      }
      // load all 8 B fragments from LDS (plain AS(3) loads -> ds_load_b128)
      union { uint4v u[2]; v16bf v; } bfr[8];
      #pragma unroll
      for (int ns = 0; ns < 8; ++ns) {
        const uint4v* bp =
            (const uint4v*)&lW[(size_t)(ns * 16 + l16) * LDW + kt + lhalf * 8];
        bfr[ns].u[0] = bp[0];
        bfr[ns].u[1] = bp[2];
      }
      #pragma unroll
      for (int ns = 0; ns < 8; ++ns)
        acc[ns] = __builtin_amdgcn_wmma_f32_16x16x32_bf16(
            false, afc.v, false, bfr[ns].v, (short)0, acc[ns], false, false);
      afc = afn;
    }

    // ---- epilogue ----
    // C layout: VGPR r holds M=r (lanes 0-15, N=lane) and M=8+r (lanes
    // 16-31, N=lane-16)  =>  row = mwbase + lhalf*8 + r, col uses l16.
    if (STORE_F32) {
      #pragma unroll
      for (int ns = 0; ns < 8; ++ns) {
        const int col = n0 + ns * 16 + l16;
        if (col < Nrows) {
          #pragma unroll
          for (int r = 0; r < 8; ++r) {
            const int row = mwbase + lhalf * 8 + r;
            Out[(size_t)row * ldOut + col] = acc[ns][r];
          }
        }
      }
    }

    if (STATS) {
      #pragma unroll
      for (int r = 0; r < 8; ++r) {
        const int row  = mwbase + lhalf * 8 + r;
        const int trow = row + shiftK;
        const int tcol = (trow < S_LEN) ? targets[trow] : -1;
        float se = 0.0f, tv = 0.0f;
        bool  thit = false;
        #pragma unroll
        for (int ns = 0; ns < 8; ++ns) {
          const int col = n0 + ns * 16 + l16;
          const float v = acc[ns][r];
          if (col < Nrows) se += __expf(v);
          if (col == tcol) { tv = v; thit = true; }
        }
        // reduce across the 16 lanes of this half-wave (masks < 16 keep the
        // two halves, which hold different rows, independent)
        #pragma unroll
        for (int m = 8; m >= 1; m >>= 1) se += __shfl_xor(se, m, 32);
        if (l16 == 0) atomicAdd(&sumexp[row], se);   // global_atomic_add_f32
        if (thit) tgtLogit[row] = tv;                 // unique writer
      }
    }
  }
}

// ---------------------------------------------------------------------------
// Final loss: per head h (shift k = h+1):
//   nll[s] = log(sumexp[h][s]) - tgt[h][s]   for s+k < S && targets[s+k]!=-100
//   loss = mean(nll_main) + 0.3 * mean_h(mean(nll_aux_h))
// ---------------------------------------------------------------------------
__global__ __launch_bounds__(256)
void loss_kernel(const float* __restrict__ sumexp,
                 const float* __restrict__ tgtl,
                 const int* __restrict__ targets,
                 float* __restrict__ out_loss) {
  __shared__ float sh[8];
  __shared__ float hsum[3], hcnt[3];
  const int wv = threadIdx.x >> 5, ln = threadIdx.x & 31;

  for (int h = 0; h < 3; ++h) {
    const int k = h + 1;
    float ls = 0.0f, lc = 0.0f;
    for (int s = threadIdx.x; s < S_LEN; s += blockDim.x) {
      const int tr = s + k;
      if (tr >= S_LEN) continue;
      const int t = targets[tr];
      if (t == IGNORE_IDX) continue;
      ls += __logf(sumexp[h * S_LEN + s]) - tgtl[h * S_LEN + s];
      lc += 1.0f;
    }
    #pragma unroll
    for (int m = 16; m >= 1; m >>= 1) {
      ls += __shfl_xor(ls, m, 32);
      lc += __shfl_xor(lc, m, 32);
    }
    if (ln == 0) sh[wv] = ls;
    __syncthreads();
    if (threadIdx.x == 0) { float a = 0; for (int i = 0; i < 8; ++i) a += sh[i]; hsum[h] = a; }
    __syncthreads();
    if (ln == 0) sh[wv] = lc;
    __syncthreads();
    if (threadIdx.x == 0) { float a = 0; for (int i = 0; i < 8; ++i) a += sh[i]; hcnt[h] = a; }
    __syncthreads();
  }
  if (threadIdx.x == 0) {
    const float main_l = hsum[0] / fmaxf(hcnt[0], 1.0f);
    const float aux1   = hsum[1] / fmaxf(hcnt[1], 1.0f);
    const float aux2   = hsum[2] / fmaxf(hcnt[2], 1.0f);
    *out_loss = main_l + LOSS_W * 0.5f * (aux1 + aux2);
  }
}

// ---------------------------------------------------------------------------
extern "C" void kernel_launch(void* const* d_in, const int* in_sizes, int n_in,
                              void* d_out, int out_size, void* d_ws, size_t ws_size,
                              hipStream_t stream) {
  const float* hidden  = (const float*)d_in[0];                  // (S,D)
  const int*   targets = (const int*)d_in[1];                    // (S,)
  const float* emb_w   = (const float*)d_in[2];                  // (V,D)
  const float* main_nw = (const float*)d_in[3];                  // (D,)
  const float* aux_pw  = (const float*)d_in[4];                  // (2,D,D)
  const float* aux_nw  = (const float*)d_in[5];                  // (2,D)
  const float* aux_ow  = (const float*)d_in[6];                  // (2,V,D)

  float* out  = (float*)d_out;
  float* loss = out + (size_t)S_LEN * V_DIM;                     // last element

  // workspace carve-out (~24.5 MB)
  char* ws = (char*)d_ws;
  __bf16* Amain = (__bf16*)ws;  ws += (size_t)S_LEN * D_DIM * sizeof(__bf16);
  __bf16* Hraw  = (__bf16*)ws;  ws += (size_t)S_LEN * D_DIM * sizeof(__bf16);
  __bf16* Aaux0 = (__bf16*)ws;  ws += (size_t)S_LEN * D_DIM * sizeof(__bf16);
  __bf16* Aaux1 = (__bf16*)ws;  ws += (size_t)S_LEN * D_DIM * sizeof(__bf16);
  float*  Htmp  = (float*)ws;   ws += (size_t)S_LEN * D_DIM * sizeof(float);
  float*  sumexp= (float*)ws;   ws += (size_t)3 * S_LEN * sizeof(float);
  float*  tgtl  = (float*)ws;   ws += (size_t)3 * S_LEN * sizeof(float);

  // allow the 258 KB dynamic-LDS slab (CDNA5: 320 KB per WGP/workgroup)
  (void)hipFuncSetAttribute((const void*)gemm_wslab_wmma_kernel<true,  false>,
                            hipFuncAttributeMaxDynamicSharedMemorySize, (int)LDS_BYTES);
  (void)hipFuncSetAttribute((const void*)gemm_wslab_wmma_kernel<true,  true>,
                            hipFuncAttributeMaxDynamicSharedMemorySize, (int)LDS_BYTES);
  (void)hipFuncSetAttribute((const void*)gemm_wslab_wmma_kernel<false, true>,
                            hipFuncAttributeMaxDynamicSharedMemorySize, (int)LDS_BYTES);

  init_stats_kernel<<<(3 * S_LEN + 255) / 256, 256, 0, stream>>>(sumexp, tgtl, 3 * S_LEN);

  // main-head norm (-> Amain) + raw bf16 copy of hidden (-> Hraw)
  rmsnorm_bf16_kernel<<<S_LEN, 256, 0, stream>>>(hidden, main_nw, Amain, Hraw);

  // proj: W tiny (4 MB) -> spread M-blocks across grid.y for parallelism
  const dim3 gProj(D_DIM / 128, 16);
  // vocab: grid.y=2 -> weights streamed from HBM only twice (~1.2 GB total)
  const dim3 gVoc((V_DIM + 127) / 128, 2);
  __bf16* Aaux[2] = {Aaux0, Aaux1};

  // aux heads: projection GEMM (bf16 WMMA) -> rmsnorm -> bf16 activations
  for (int i = 0; i < 2; ++i) {
    gemm_wslab_wmma_kernel<true, false><<<gProj, 256, LDS_BYTES, stream>>>(
        Hraw, aux_pw + (size_t)i * D_DIM * D_DIM, D_DIM,
        Htmp, (long long)D_DIM, nullptr, nullptr, nullptr, 0);
    rmsnorm_bf16_kernel<<<S_LEN, 256, 0, stream>>>(
        Htmp, aux_nw + (size_t)i * D_DIM, Aaux[i], nullptr);
  }

  // main head: store logits to d_out + fused CE stats (shift k = 1)
  gemm_wslab_wmma_kernel<true, true><<<gVoc, 256, LDS_BYTES, stream>>>(
      Amain, emb_w, V_DIM, out, (long long)V_DIM,
      sumexp + 0 * S_LEN, tgtl + 0 * S_LEN, targets, 1);

  // aux heads: CE stats only, logits never materialized (shift k = 2, 3)
  for (int i = 0; i < 2; ++i) {
    gemm_wslab_wmma_kernel<false, true><<<gVoc, 256, LDS_BYTES, stream>>>(
        Aaux[i], aux_ow + (size_t)i * V_DIM * D_DIM, V_DIM,
        nullptr, 0,
        sumexp + (size_t)(i + 1) * S_LEN, tgtl + (size_t)(i + 1) * S_LEN,
        targets, i + 2);
  }

  loss_kernel<<<1, 256, 0, stream>>>(sumexp, tgtl, targets, loss);
}
